// SerializedConvolution_53171695125273
// MI455X (gfx1250) — matrix operand order repaired and checked by reading, not compiled, not run on hardware
//
#include <hip/hip_runtime.h>
#include <hip/hip_bf16.h>
#include <stdint.h>

// Problem dims (fixed by the reference harness)
#define BB 8
#define NN 16384
#define CC 256
#define FF 1024
#define OO 256
#define DEPTH_EFF 18   // score_scale floor-divides to 0 after iter 17; later iters don't affect argsort

typedef __attribute__((ext_vector_type(16))) __bf16 v16bf;
typedef __attribute__((ext_vector_type(8)))  float  v8f;
typedef __attribute__((ext_vector_type(4)))  unsigned int v4u;
typedef __attribute__((ext_vector_type(8)))  int    v8i;
typedef __attribute__((ext_vector_type(4)))  int    v4i;

// ---------------------------------------------------------------------------
// WMMA helpers (bf16 16x16x32; A and B both use contiguous k = 16*half + e)
// ---------------------------------------------------------------------------
__device__ __forceinline__ v8f wmma_bf16(v16bf a, v16bf b, v8f c) {
  return __builtin_amdgcn_wmma_f32_16x16x32_bf16(false, a, false, b, (short)0, c, false, false);
}

// A fragment from a bf16 row-major weight row (global or LDS):
// lane holds row M=lane%16, half=lane/16.
// elements e=0..15 -> K = kbase + 16*half + e
// 16 contiguous elements -> 2 x b128 loads.
__device__ __forceinline__ v16bf a_frag_from_bf16(const __bf16* Wrow, int kbase, int hh) {
  union { uint4 u[2]; v16bf v; } tmp;
  tmp.u[0] = *(const uint4*)(Wrow + kbase + 16 * hh);
  tmp.u[1] = *(const uint4*)(Wrow + kbase + 16 * hh + 8);
  return tmp.v;
}

// B fragment: lane holds col N=lane%16, half=lane/16; K = koff + e (16 contiguous).
// rowbase points at an n-major LDS tile row (16B-aligned at koff).
__device__ __forceinline__ v16bf b_frag_from_lds(const __bf16* rowbase, int koff) {
  union { uint4 u[2]; v16bf v; } tmp;
  tmp.u[0] = *(const uint4*)(rowbase + koff);
  tmp.u[1] = *(const uint4*)(rowbase + koff + 8);
  return tmp.v;
}

// Stage 8 bf16 (one coalesced b128 global load worth, along n) into an
// n-major LDS tile at column kk: 8 x ds_store_b16 scatter (transpose).
__device__ __forceinline__ void stage_row8(__bf16* ldsbase, int ldsstride, int nloc, int kk,
                                           uint4 v) {
  union { uint4 u; __bf16 b[8]; } t;
  t.u = v;
#pragma unroll
  for (int i = 0; i < 8; ++i) ldsbase[(nloc + i) * ldsstride + kk] = t.b[i];
}

// ---------------------------------------------------------------------------
// TDM: 2D tile -> LDS (D# per ISA 8.3/8.4). dscode: 0=1B,1=2B,2=4B,3=8B.
// clang>=23 exposes the 6-arg builtin, ROCm7.2 clang-22 the 5-arg form.
// ---------------------------------------------------------------------------
__device__ __forceinline__ void tdm_load_2d(unsigned lds_off, const void* gptr,
                                            unsigned dscode,
                                            unsigned tensor_w, unsigned tensor_h,
                                            unsigned stride_w,
                                            unsigned tile_w, unsigned tile_h) {
  unsigned long long ga = (unsigned long long)(uintptr_t)gptr;
  v4u g0;
  g0[0] = 1u;                                               // count=1, user mode
  g0[1] = lds_off;                                          // lds_addr (bytes)
  g0[2] = (unsigned)(ga & 0xFFFFFFFFu);                     // global_addr[31:0]
  g0[3] = (unsigned)((ga >> 32) & 0x1FFFFFFu) | (2u << 30); // addr[56:32], type=2
  v8i g1;
  g1[0] = (int)(dscode << 16);                              // data_size, wg_mask=0
  g1[1] = (int)((tensor_w & 0xFFFFu) << 16);                // tensor_dim0[15:0]
  g1[2] = (int)(((tensor_w >> 16) & 0xFFFFu) | ((tensor_h & 0xFFFFu) << 16));
  g1[3] = (int)(((tensor_h >> 16) & 0xFFFFu) | ((tile_w & 0xFFFFu) << 16));
  g1[4] = (int)(tile_h & 0xFFFFu);                          // tile_dim1 (tile_dim2=0)
  g1[5] = (int)stride_w;                                    // tensor_dim0_stride[31:0]
  g1[6] = 0;
  g1[7] = 0;
  v4i z4 = {0, 0, 0, 0};
#if defined(__clang_major__) && (__clang_major__ >= 23)
  v8i z8 = {0, 0, 0, 0, 0, 0, 0, 0};
  __builtin_amdgcn_tensor_load_to_lds(g0, g1, z4, z4, z8, 0);
#else
  __builtin_amdgcn_tensor_load_to_lds(g0, g1, z4, z4, 0);
#endif
}

// ---------------------------------------------------------------------------
// threefry2x32 PRNG (deterministic; structure-faithful to the reference)
// ---------------------------------------------------------------------------
__device__ __forceinline__ unsigned rotl32(unsigned x, int r) { return (x << r) | (x >> (32 - r)); }

__device__ __forceinline__ void threefry(unsigned k0, unsigned k1, unsigned c0, unsigned c1,
                                         unsigned& o0, unsigned& o1) {
  const unsigned ks2 = k0 ^ k1 ^ 0x1BD11BDAu;
  unsigned x0 = c0 + k0, x1 = c1 + k1;
  const int RA[4] = {13, 15, 26, 6};
  const int RB[4] = {17, 29, 16, 24};
#pragma unroll
  for (int j = 0; j < 4; ++j) { x0 += x1; x1 = rotl32(x1, RA[j]); x1 ^= x0; }
  x0 += k1; x1 += ks2 + 1;
#pragma unroll
  for (int j = 0; j < 4; ++j) { x0 += x1; x1 = rotl32(x1, RB[j]); x1 ^= x0; }
  x0 += ks2; x1 += k0 + 2;
#pragma unroll
  for (int j = 0; j < 4; ++j) { x0 += x1; x1 = rotl32(x1, RA[j]); x1 ^= x0; }
  x0 += k0; x1 += k1 + 3;
#pragma unroll
  for (int j = 0; j < 4; ++j) { x0 += x1; x1 = rotl32(x1, RB[j]); x1 ^= x0; }
  x0 += k1; x1 += ks2 + 4;
#pragma unroll
  for (int j = 0; j < 4; ++j) { x0 += x1; x1 = rotl32(x1, RA[j]); x1 ^= x0; }
  x0 += ks2; x1 += k0 + 5;
  o0 = x0; o1 = x1;
}

__device__ __forceinline__ float u01(unsigned r) { return (float)(r >> 8) * (1.0f / 16777216.0f); }

__device__ __forceinline__ float gauss(unsigned c0, unsigned c1) {
  unsigned r0, r1;
  threefry(0x2Au, 0x9E3779B9u, c0, c1, r0, r1);
  float u1 = ((float)(r0 >> 8) + 0.5f) * (1.0f / 16777216.0f);
  float u2 = u01(r1);
  return sqrtf(-2.0f * logf(u1)) * __cosf(6.28318530718f * u2);
}

// ---------------------------------------------------------------------------
// Kernel 0: fp32 -> bf16 weight pre-pack (vectorized, one-shot, ~2MB total)
// ---------------------------------------------------------------------------
__global__ __launch_bounds__(256) void cvt_bf16_kernel(const float* __restrict__ src,
                                                       __bf16* __restrict__ dst, int n4) {
  int i = blockIdx.x * 256 + threadIdx.x;
  if (i < n4) {
    float4 f = ((const float4*)src)[i];
    union { __bf16 b[4]; uint2 u; } o;
    o.b[0] = (__bf16)f.x; o.b[1] = (__bf16)f.y; o.b[2] = (__bf16)f.z; o.b[3] = (__bf16)f.w;
    ((uint2*)dst)[i] = o.u;
  }
}

// ---------------------------------------------------------------------------
// Kernel 1: random serialization. One 1024-thread WG per batch; 16 pts/thread
// in registers; LDS atomics for mean & duplet argmins; u64 bitonic sort in
// 128KB dynamic LDS (key = orderedFloat(total)<<32 | index : stable argsort).
// ---------------------------------------------------------------------------
__global__ __launch_bounds__(1024) void ser_kernel(const float* __restrict__ coords,
                                                   const float* __restrict__ mask,
                                                   int* __restrict__ idx_out) {
  extern __shared__ char dynsmem[];
  unsigned long long* keys = (unsigned long long*)dynsmem;  // NN entries
  __shared__ float s_sum;
  __shared__ float s_cnt;
  __shared__ unsigned long long s_min[4];
  __shared__ float s_pt[4][3];

  const int b = blockIdx.x, tid = threadIdx.x;
  float px[16], py[16], pz[16], tot[16];
  unsigned mbits = 0;
#pragma unroll
  for (int r = 0; r < 16; ++r) {
    int j = tid + r * 1024;
    size_t base = ((size_t)b * NN + j) * 3;
    px[r] = coords[base]; py[r] = coords[base + 1]; pz[r] = coords[base + 2];
    tot[r] = 0.f;
    if (mask[(size_t)b * NN + j] > 0.f) mbits |= (1u << r);
  }
  if (tid == 0) s_cnt = 0.f;
  __syncthreads();
  atomicAdd(&s_cnt, (float)__popc(mbits));

  float oa = 1.f, ob = 1.f, oc = 1.f;
  for (int it = 0; it < DEPTH_EFF; ++it) {
    __syncthreads();
    if (tid == 0) s_sum = 0.f;
    if (tid < 4) s_min[tid] = ~0ull;
    __syncthreads();
    float a0 = gauss((unsigned)((b << 8) | it), 0xA0000000u);
    float a1 = gauss((unsigned)((b << 8) | it), 0xA0000001u);
    float a2 = gauss((unsigned)((b << 8) | it), 0xA0000002u);
    float dp = a0 * oa + a1 * ob + a2 * oc;
    float sg = (dp < 0.f) ? -1.f : 1.f;   // sign(0) -> +1 per reference
    a0 *= sg; a1 *= sg; a2 *= sg;
    oa = a0; ob = a1; oc = a2;

    float pr[16]; float lsum = 0.f;
#pragma unroll
    for (int r = 0; r < 16; ++r) {
      pr[r] = px[r] * a0 + py[r] * a1 + pz[r] * a2;
      if (mbits & (1u << r)) lsum += pr[r];
    }
    atomicAdd(&s_sum, lsum);
    __syncthreads();
    float mean = s_sum / (s_cnt + 1e-9f);

    unsigned sl = 0, sr = 0;
    float scale = (float)(1u << (17 - it));
#pragma unroll
    for (int r = 0; r < 16; ++r) {
      float d = pr[r] - mean;
      float s = (d > 0.f) ? 1.f : ((d < 0.f) ? -1.f : 0.f);
      tot[r] += s * scale;
      if (s < 0.f) sl |= (1u << r);
      if (s > 0.f) sr |= (1u << r);
    }
    sl &= mbits; sr &= mbits;

#pragma unroll
    for (int q = 0; q < 4; ++q) {
      unsigned side = (unsigned)(q >> 1), pick = (unsigned)(q & 1);
      unsigned smask = side ? sr : sl;
      unsigned long long best = ~0ull;
#pragma unroll
      for (int r = 0; r < 16; ++r) {
        int j = tid + r * 1024;
        unsigned r0, r1;
        threefry(0x51DEu + side * 2u + pick, 0x2Au,
                 (unsigned)((b << 8) | it), (unsigned)j, r0, r1);
        float sc = u01(r0) + (((smask >> r) & 1u) ? 0.f : 1e9f);
        unsigned long long kk = (((unsigned long long)__float_as_uint(sc)) << 32) | (unsigned)j;
        best = (kk < best) ? kk : best;
      }
      atomicMin(&s_min[q], best);
    }
    __syncthreads();
#pragma unroll
    for (int q = 0; q < 4; ++q) {
      unsigned j = (unsigned)(s_min[q] & 0xFFFFFFFFu);
      if ((j & 1023u) == (unsigned)tid) {
        int r = (int)(j >> 10);
        s_pt[q][0] = px[r]; s_pt[q][1] = py[r]; s_pt[q][2] = pz[r];
      }
    }
    __syncthreads();
    float mlx = 0.5f * (s_pt[0][0] + s_pt[1][0]), mly = 0.5f * (s_pt[0][1] + s_pt[1][1]),
          mlz = 0.5f * (s_pt[0][2] + s_pt[1][2]);
    float mrx = 0.5f * (s_pt[2][0] + s_pt[3][0]), mry = 0.5f * (s_pt[2][1] + s_pt[3][1]),
          mrz = 0.5f * (s_pt[2][2] + s_pt[3][2]);
#pragma unroll
    for (int r = 0; r < 16; ++r) {
      bool l = (sl >> r) & 1u, rr = (sr >> r) & 1u;
      float nx = l ? px[r] - mlx : (rr ? px[r] - mrx : 0.f);
      float ny = l ? py[r] - mly : (rr ? py[r] - mry : 0.f);
      float nz = l ? pz[r] - mlz : (rr ? pz[r] - mrz : 0.f);
      px[r] = nx; py[r] = ny; pz[r] = nz;
    }
  }

#pragma unroll
  for (int r = 0; r < 16; ++r) {
    int j = tid + r * 1024;
    float f = tot[r] + 524288.0f * (((mbits >> r) & 1u) ? 0.f : 1.f);  // inf = 2^19
    unsigned u = __float_as_uint(f);
    u = (u & 0x80000000u) ? ~u : (u | 0x80000000u);
    keys[j] = (((unsigned long long)u) << 32) | (unsigned)j;
  }

  for (unsigned k = 2; k <= (unsigned)NN; k <<= 1) {
    for (unsigned s = k >> 1; s > 0; s >>= 1) {
      __syncthreads();
      for (unsigned t = tid; t < (unsigned)NN / 2; t += 1024u) {
        unsigned lo = t & (s - 1);
        unsigned i = ((t ^ lo) << 1) | lo;
        unsigned j2 = i | s;
        unsigned long long A = keys[i], Bv = keys[j2];
        bool asc = ((i & k) == 0);
        bool sw = asc ? (A > Bv) : (A < Bv);
        if (sw) { keys[i] = Bv; keys[j2] = A; }
      }
    }
  }
  __syncthreads();
  for (int p = tid; p < NN; p += 1024)
    idx_out[(size_t)b * NN + p] = (int)(keys[p] & 0xFFFFFFFFu);
}

// ---------------------------------------------------------------------------
// Kernel 2: gather + mask -> xs bf16 (B,C,N)
// ---------------------------------------------------------------------------
__global__ __launch_bounds__(256) void gather_kernel(const float* __restrict__ x,
                                                     const float* __restrict__ mask,
                                                     const int* __restrict__ sidx,
                                                     __bf16* __restrict__ XS) {
  __shared__ int s_src[64];
  __shared__ float s_m[64];
  const int b = blockIdx.y, n0 = blockIdx.x * 64, c = threadIdx.x;
  if (c < 64) {
    s_src[c] = sidx[(size_t)b * NN + n0 + c];
    s_m[c]   = mask[(size_t)b * NN + n0 + c];
  }
  __syncthreads();
  __bf16* dst = XS + ((size_t)b * CC + c) * NN + n0;
#pragma unroll 8
  for (int r = 0; r < 64; ++r)
    dst[r] = (__bf16)(x[((size_t)b * NN + s_src[r]) * CC + c] * s_m[r]);
}

// ---------------------------------------------------------------------------
// Kernel 3: depthwise conv (K=9, zero-pad) + bias + ConvRMSNorm -> y bf16
// ---------------------------------------------------------------------------
__global__ __launch_bounds__(256) void dwconv_kernel(const __bf16* __restrict__ XS,
                                                     const float* __restrict__ w1,
                                                     const float* __restrict__ b1,
                                                     __bf16* __restrict__ Y) {
  extern __shared__ char dynsmem[];
  float* ytile = (float*)dynsmem;  // [256][65] padded vs bank conflicts
  __shared__ float colss[64];
  const int b = blockIdx.y, n0 = blockIdx.x * 64, c = threadIdx.x;
  if (c < 64) colss[c] = 0.f;

  float w[9];
#pragma unroll
  for (int k = 0; k < 9; ++k) w[k] = w1[c * 9 + k];
  const float bias = b1[c];
  const __bf16* row = XS + ((size_t)b * CC + c) * NN;

  float win[9];
#pragma unroll
  for (int k = 0; k < 9; ++k) {
    int n = n0 - 4 + k;
    win[k] = (n >= 0 && n < NN) ? (float)row[n] : 0.f;
  }
  float* yr = ytile + c * 65;
  for (int o = 0; o < 64; ++o) {
    float acc = bias;
#pragma unroll
    for (int k = 0; k < 9; ++k) acc += win[k] * w[k];
    yr[o] = acc;
#pragma unroll
    for (int k = 0; k < 8; ++k) win[k] = win[k + 1];
    int n = n0 + 5 + o;
    win[8] = (n < NN) ? (float)row[n] : 0.f;
  }
  __syncthreads();
  {
    int o = threadIdx.x & 63, g = threadIdx.x >> 6;
    float ss = 0.f;
    for (int cc = g * 64; cc < g * 64 + 64; ++cc) {
      float v = ytile[cc * 65 + o];
      ss += v * v;
    }
    atomicAdd(&colss[o], ss);
  }
  __syncthreads();
  __bf16* dst = Y + ((size_t)b * CC + c) * NN + n0;
#pragma unroll
  for (int o = 0; o < 64; ++o) {
    float rms = sqrtf(colss[o]) * 0.0625f;  // ||y||/sqrt(256)
    dst[o] = (__bf16)(yr[o] / (rms + 1e-8f));
  }
}

// ---------------------------------------------------------------------------
// Kernel 4: xb = conv2_w @ y + b2 -> bf16 (B,F,N). Block = 64M x 64N, 8 waves
// (4x2), each wave: 1 A-frag + 2 B-frags -> 2 WMMA per k-step. bf16 weight
// block (64x256 = 32KB) staged once by TDM. x-tile staged with one b128
// global load per thread per k-step.
// ---------------------------------------------------------------------------
__global__ __launch_bounds__(256) void gemm_xb_kernel(const __bf16* __restrict__ Y,
                                                      const __bf16* __restrict__ W2b,
                                                      const float* __restrict__ b2,
                                                      __bf16* __restrict__ XB) {
  extern __shared__ char dynsmem[];
  __bf16* Wl = (__bf16*)dynsmem;  // [64][256] bf16, 32KB, TDM destination
  __shared__ __bf16 Xt[64][40];   // n-major y tile, stride 40 keeps 16B align
  const int b = blockIdx.z, m0 = blockIdx.x * 64, n0 = blockIdx.y * 64;
  const int wave = threadIdx.x >> 5, lane = threadIdx.x & 31;
  const int wm = wave & 3, wn = wave >> 2, hh = lane >> 4, l16 = lane & 15;
  const int skk = threadIdx.x >> 3, sng = (threadIdx.x & 7) * 8;  // staging map

  if (wave == 0) {
    tdm_load_2d((unsigned)(uintptr_t)Wl, W2b + (size_t)m0 * CC, 1u /*2B*/,
                CC, (unsigned)(FF - m0), CC, CC, 64);
    __builtin_amdgcn_s_wait_tensorcnt((short)0);
  }

  v8f acc0 = {0.f, 0.f, 0.f, 0.f, 0.f, 0.f, 0.f, 0.f};
  v8f acc1 = {0.f, 0.f, 0.f, 0.f, 0.f, 0.f, 0.f, 0.f};
  const __bf16* wrow = Wl + (wm * 16 + l16) * CC;
  for (int k0 = 0; k0 < CC; k0 += 32) {
    __syncthreads();
    {
      uint4 v = *(const uint4*)(Y + ((size_t)b * CC + k0 + skk) * NN + n0 + sng);
      stage_row8(&Xt[0][0], 40, sng, skk, v);
    }
    __syncthreads();
    v16bf a = a_frag_from_bf16(wrow, k0, hh);
    v16bf bf0 = b_frag_from_lds(&Xt[wn * 32 + l16][0], 16 * hh);
    v16bf bf1 = b_frag_from_lds(&Xt[wn * 32 + 16 + l16][0], 16 * hh);
    acc0 = wmma_bf16(a, bf0, acc0);
    acc1 = wmma_bf16(a, bf1, acc1);
  }
#pragma unroll
  for (int sub = 0; sub < 2; ++sub) {
    int col = n0 + wn * 32 + sub * 16 + l16;
    const v8f& acc = sub ? acc1 : acc0;
#pragma unroll
    for (int r = 0; r < 8; ++r) {
      int m = m0 + wm * 16 + r + 8 * hh;
      XB[((size_t)b * FF + m) * NN + col] = (__bf16)(acc[r] + b2[m]);
    }
  }
}

// ---------------------------------------------------------------------------
// Kernel 5: att = sigmoid(mean_c((sm_w@xs + smb) * (sb_w@xb + sbb))).
// Block covers 16 columns; A-frags straight from pre-packed bf16 weights;
// x tiles staged with b128 global loads (2 n-groups x 128 k-rows per pass).
// ---------------------------------------------------------------------------
__global__ __launch_bounds__(256) void att_kernel(const __bf16* __restrict__ XS,
                                                  const __bf16* __restrict__ XB,
                                                  const __bf16* __restrict__ smwb,
                                                  const float* __restrict__ smb,
                                                  const __bf16* __restrict__ sbwb,
                                                  const float* __restrict__ sbb,
                                                  float* __restrict__ att) {
  __shared__ __bf16 XsT[16][264];   // n-major, stride mult of 8 -> 16B align
  __shared__ __bf16 XbT[16][1032];
  __shared__ float colsum[16];
  const int b = blockIdx.y, n0 = blockIdx.x * 16;
  const int wave = threadIdx.x >> 5, lane = threadIdx.x & 31;
  const int hh = lane >> 4, l16 = lane & 15;
  if (threadIdx.x < 16) colsum[threadIdx.x] = 0.f;
  {
    int nn8 = (threadIdx.x & 1) * 8, kk0 = threadIdx.x >> 1;  // 128 k-rows/pass
#pragma unroll
    for (int kk = kk0; kk < CC; kk += 128) {
      uint4 v = *(const uint4*)(XS + ((size_t)b * CC + kk) * NN + n0 + nn8);
      stage_row8(&XsT[0][0], 264, nn8, kk, v);
    }
#pragma unroll
    for (int kk = kk0; kk < FF; kk += 128) {
      uint4 v = *(const uint4*)(XB + ((size_t)b * FF + kk) * NN + n0 + nn8);
      stage_row8(&XbT[0][0], 1032, nn8, kk, v);
    }
  }
  __syncthreads();

  float part = 0.f;
  for (int mi = wave; mi < 16; mi += 8) {
    int mt = mi * 16;
    v8f accA = {0.f, 0.f, 0.f, 0.f, 0.f, 0.f, 0.f, 0.f};
    v8f accB = {0.f, 0.f, 0.f, 0.f, 0.f, 0.f, 0.f, 0.f};
    const __bf16* warow = smwb + (size_t)(mt + l16) * CC;
    for (int k0 = 0; k0 < CC; k0 += 32)
      accA = wmma_bf16(a_frag_from_bf16(warow, k0, hh),
                       b_frag_from_lds(&XsT[l16][0], k0 + 16 * hh), accA);
    const __bf16* wbrow = sbwb + (size_t)(mt + l16) * FF;
    for (int k0 = 0; k0 < FF; k0 += 32)
      accB = wmma_bf16(a_frag_from_bf16(wbrow, k0, hh),
                       b_frag_from_lds(&XbT[l16][0], k0 + 16 * hh), accB);
#pragma unroll
    for (int r = 0; r < 8; ++r) {
      int m = mt + r + 8 * hh;
      part += (accA[r] + smb[m]) * (accB[r] + sbb[m]);
    }
  }
  atomicAdd(&colsum[l16], part);
  __syncthreads();
  if (threadIdx.x < 16) {
    float v = colsum[threadIdx.x] * (1.0f / 256.0f);
    att[(size_t)b * NN + n0 + threadIdx.x] = 1.0f / (1.0f + __expf(-v));
  }
}

// ---------------------------------------------------------------------------
// Kernel 6: conv3 @ gelu(xb) + b3, fused att*scale + masked residual and
// de-serialization scatter to d_out (B,N,O). Block = 64M x 64N; the 64x1024
// bf16 conv3 block (128KB) is TDM-staged into LDS once; gelu (exact erf form)
// applied while staging the xb tile (one b128 global load per thread/step).
// ---------------------------------------------------------------------------
__global__ __launch_bounds__(256) void out_kernel(const __bf16* __restrict__ XB,
                                                  const __bf16* __restrict__ W3b,
                                                  const float* __restrict__ b3,
                                                  const float* __restrict__ att,
                                                  const int* __restrict__ sidx,
                                                  const float* __restrict__ mask,
                                                  const float* __restrict__ x,
                                                  const int* __restrict__ rscale,
                                                  float* __restrict__ out) {
  extern __shared__ char dynsmem[];
  __bf16* Wl = (__bf16*)dynsmem;  // [64][1024] bf16, 128KB, TDM destination
  __shared__ __bf16 Xt[64][40];
  const int b = blockIdx.z, m0 = blockIdx.x * 64, n0 = blockIdx.y * 64;
  const int wave = threadIdx.x >> 5, lane = threadIdx.x & 31;
  const int wm = wave & 3, wn = wave >> 2, hh = lane >> 4, l16 = lane & 15;
  const int skk = threadIdx.x >> 3, sng = (threadIdx.x & 7) * 8;

  if (wave == 0) {
    tdm_load_2d((unsigned)(uintptr_t)Wl, W3b + (size_t)m0 * FF, 1u /*2B*/,
                FF, (unsigned)(OO - m0), FF, FF, 64);
    __builtin_amdgcn_s_wait_tensorcnt((short)0);
  }

  v8f acc0 = {0.f, 0.f, 0.f, 0.f, 0.f, 0.f, 0.f, 0.f};
  v8f acc1 = {0.f, 0.f, 0.f, 0.f, 0.f, 0.f, 0.f, 0.f};
  const __bf16* wrow = Wl + (wm * 16 + l16) * FF;
  for (int k0 = 0; k0 < FF; k0 += 32) {
    __syncthreads();
    {
      union { uint4 u; __bf16 bfv[8]; } t;
      t.u = *(const uint4*)(XB + ((size_t)b * FF + k0 + skk) * NN + n0 + sng);
#pragma unroll
      for (int i = 0; i < 8; ++i) {
        float v = (float)t.bfv[i];
        Xt[sng + i][skk] = (__bf16)(0.5f * v * (1.0f + erff(v * 0.70710678118f)));
      }
    }
    __syncthreads();
    v16bf a = a_frag_from_bf16(wrow, k0, hh);
    acc0 = wmma_bf16(a, b_frag_from_lds(&Xt[wn * 32 + l16][0], 16 * hh), acc0);
    acc1 = wmma_bf16(a, b_frag_from_lds(&Xt[wn * 32 + 16 + l16][0], 16 * hh), acc1);
  }
#pragma unroll
  for (int sub = 0; sub < 2; ++sub) {
    int col = n0 + wn * 32 + sub * 16 + l16;
    const v8f& acc = sub ? acc1 : acc0;
    int src = sidx[(size_t)b * NN + col];
    float sc = att[(size_t)b * NN + col] * (float)rscale[0];
    float mk = mask[(size_t)b * NN + col];
    const float* xr = x + ((size_t)b * NN + src) * CC;  // O == C, residual valid
    float* orow = out + ((size_t)b * NN + src) * OO;
#pragma unroll
    for (int r = 0; r < 8; ++r) {
      int m = m0 + wm * 16 + r + 8 * hh;
      orow[m] = xr[m] * mk + (acc[r] + b3[m]) * sc;
    }
  }
}

// ---------------------------------------------------------------------------
// Launch
// ---------------------------------------------------------------------------
extern "C" void kernel_launch(void* const* d_in, const int* in_sizes, int n_in,
                              void* d_out, int out_size, void* d_ws, size_t ws_size,
                              hipStream_t stream) {
  const float* x      = (const float*)d_in[0];
  const float* coords = (const float*)d_in[1];
  const float* mask   = (const float*)d_in[2];
  const int*   rscale = (const int*)d_in[3];
  const float* w1     = (const float*)d_in[4];
  const float* b1     = (const float*)d_in[5];
  const float* w2     = (const float*)d_in[6];
  const float* b2     = (const float*)d_in[7];
  const float* smw    = (const float*)d_in[8];
  const float* smb    = (const float*)d_in[9];
  const float* sbw    = (const float*)d_in[10];
  const float* sbb    = (const float*)d_in[11];
  const float* w3     = (const float*)d_in[12];
  const float* b3     = (const float*)d_in[13];
  float* out = (float*)d_out;

  // workspace: idx | att | xs | y | xb | bf16 weights  (~387MB)
  char* ws = (char*)d_ws;
  int*    idx    = (int*)ws;
  float*  attbuf = (float*)(ws + (size_t)BB * NN * 4);
  __bf16* xs     = (__bf16*)(ws + (size_t)2 * BB * NN * 4);
  __bf16* yb     = xs + (size_t)BB * CC * NN;
  __bf16* xb     = yb + (size_t)BB * CC * NN;
  __bf16* wb2    = xb + (size_t)BB * FF * NN;
  __bf16* wsm    = wb2 + (size_t)FF * CC;
  __bf16* wsb    = wsm + (size_t)CC * CC;
  __bf16* ww3    = wsb + (size_t)CC * FF;

  cvt_bf16_kernel<<<(FF * CC / 4 + 255) / 256, 256, 0, stream>>>(w2, wb2, FF * CC / 4);
  cvt_bf16_kernel<<<(CC * CC / 4 + 255) / 256, 256, 0, stream>>>(smw, wsm, CC * CC / 4);
  cvt_bf16_kernel<<<(CC * FF / 4 + 255) / 256, 256, 0, stream>>>(sbw, wsb, CC * FF / 4);
  cvt_bf16_kernel<<<(OO * FF / 4 + 255) / 256, 256, 0, stream>>>(w3, ww3, OO * FF / 4);

  ser_kernel<<<BB, 1024, NN * sizeof(unsigned long long), stream>>>(coords, mask, idx);
  gather_kernel<<<dim3(NN / 64, BB), 256, 0, stream>>>(x, mask, idx, xs);
  dwconv_kernel<<<dim3(NN / 64, BB), 256, 256 * 65 * sizeof(float), stream>>>(xs, w1, b1, yb);
  gemm_xb_kernel<<<dim3(FF / 64, NN / 64, BB), 256, 64 * CC * sizeof(__bf16), stream>>>(yb, wb2, b2, xb);
  att_kernel<<<dim3(NN / 16, BB), 256, 0, stream>>>(xs, xb, wsm, smb, wsb, sbb, attbuf);
  out_kernel<<<dim3(OO / 64, NN / 64, BB), 256, 64 * FF * sizeof(__bf16), stream>>>(
      xb, ww3, b3, attbuf, idx, mask, x, rscale, out);
}
